// AMsoftmax_59167469470120
// MI455X (gfx1250) — matrix operand order repaired and checked by reading, not compiled, not run on hardware
//
#include <hip/hip_runtime.h>

// ---------- problem constants ----------
#define N_ROWS   2048
#define D_DIM    192
#define C_CLS    100000
#define CT       3125        // column tiles of 32 (100000/32)
#define NCHUNK   32          // C-chunks across grid.x
#define M_TILE   128         // rows per workgroup (8 waves x 16)
#define BSTRIDE  200         // padded LDS row stride (bf16) for the f32-fallback kernel
#define TSTRIDE  216         // TDM-padded LDS row stride (bf16): 108 dwords, conflict-free
#define TDM_TILE_BYTES (32 * TSTRIDE * 2)   // 13824 B per buffer
#define K2F      43.28085122666891f   // S * log2(e) = 30/ln(2)
#define LN2F     0.6931471805599453f

typedef __attribute__((ext_vector_type(16))) __bf16 v16bf;
typedef __attribute__((ext_vector_type(8)))  __bf16 v8bf;
typedef __attribute__((ext_vector_type(4)))  __bf16 v4bf;
typedef __attribute__((ext_vector_type(8)))  float  v8f;
typedef __attribute__((ext_vector_type(4)))  unsigned int v4u;
typedef __attribute__((ext_vector_type(8)))  int v8i;
typedef __attribute__((ext_vector_type(4)))  int v4i;

union V16 { v16bf v; v8bf h[2]; };

// ---------------------------------------------------------------------------
// Kernel 1: row-normalize x, build A = xn * (S*log2e) in bf16,
//           and exact f32 target logit tgt[n] = xn . W[label[n]]
// ---------------------------------------------------------------------------
__global__ __launch_bounds__(256) void normalize_kernel(
    const float* __restrict__ x, const float* __restrict__ W,
    const int* __restrict__ label, __bf16* __restrict__ A,
    float* __restrict__ tgt)
{
    const int lane = threadIdx.x & 31;
    const int wv   = threadIdx.x >> 5;
    const int row  = blockIdx.x * 8 + wv;

    float xv[6];
#pragma unroll
    for (int j = 0; j < 6; ++j) xv[j] = x[(size_t)row * D_DIM + lane + 32 * j];

    float ss = 0.f;
#pragma unroll
    for (int j = 0; j < 6; ++j) ss += xv[j] * xv[j];
#pragma unroll
    for (int m = 16; m >= 1; m >>= 1) ss += __shfl_xor(ss, m, 32);

    float inv = rsqrtf(ss);
    inv = inv * (1.5f - 0.5f * ss * inv * inv);

    const int lab = label[row];
    const float* Wr = W + (size_t)lab * D_DIM;
    float dv = 0.f;
#pragma unroll
    for (int j = 0; j < 6; ++j) dv += xv[j] * Wr[lane + 32 * j];
#pragma unroll
    for (int m = 16; m >= 1; m >>= 1) dv += __shfl_xor(dv, m, 32);

    if (lane == 0) tgt[row] = inv * dv;

#pragma unroll
    for (int j = 0; j < 6; ++j)
        A[(size_t)row * D_DIM + lane + 32 * j] = (__bf16)(xv[j] * inv * K2F);
}

// ---------------------------------------------------------------------------
// Kernel 1b: one-shot W f32 -> bf16 (row-major, unpadded) for the TDM path
// ---------------------------------------------------------------------------
__global__ __launch_bounds__(256) void convert_w_kernel(
    const float* __restrict__ W, __bf16* __restrict__ Wb)
{
    const size_t i = (size_t)blockIdx.x * 256 + threadIdx.x;   // one float4 each
    float4 v = ((const float4*)W)[i];
    v4bf o;
    o[0] = (__bf16)v.x; o[1] = (__bf16)v.y; o[2] = (__bf16)v.z; o[3] = (__bf16)v.w;
    ((v4bf*)Wb)[i] = o;
}

// ---------------------------------------------------------------------------
// Kernel 2a (TDM path): fused GEMM + sum(exp2(logit)) over a C-chunk.
// B tiles (32x192 bf16) are DMA'd into double-buffered LDS by the Tensor Data
// Mover (one tensor_load_to_lds per tile, wave 0), with D# LDS padding of
// 4 dwords every 32 dwords -> 108-dword rows -> conflict-free ds_load_b128.
// ---------------------------------------------------------------------------
__global__ __launch_bounds__(256) void gemm_lse_tdm_kernel(
    const __bf16* __restrict__ A, const __bf16* __restrict__ Wb,
    float* __restrict__ partial)
{
    __shared__ __bf16 Bsh[2][32][TSTRIDE];   // 27.6 KB

    const int tid     = threadIdx.x;
    const int lane    = tid & 31;
    const int wv      = tid >> 5;
    const int g       = lane >> 4;
    const int ncol    = lane & 15;
    const int bx      = blockIdx.x;
    const int rowbase = blockIdx.y * M_TILE;

    // A fragments persist in registers for the whole C loop
    V16 afrag[6];
    {
        const __bf16* Ar = A + (size_t)(rowbase + 16 * wv + ncol) * D_DIM;
#pragma unroll
        for (int kk = 0; kk < 6; ++kk) {
            afrag[kk].h[0] = *(const v8bf*)(Ar + kk * 32 + 16 * g);
            afrag[kk].h[1] = *(const v8bf*)(Ar + kk * 32 + 16 * g + 8);
        }
    }

    const int t0 = (CT * bx) / NCHUNK;
    const int t1 = (CT * (bx + 1)) / NCHUNK;

    // LDS flat addresses carry the byte offset in bits[31:0] (ISA 10.2)
    const unsigned lbase = (unsigned)(unsigned long long)(void*)&Bsh[0][0][0];
    const unsigned long long wbase = (unsigned long long)Wb;

    auto tdm_issue = [&](int t, int b) {
        const unsigned long long ga = wbase + (unsigned long long)t * (32u * D_DIM * 2u);
        // D# group0: count=1 | lds_addr | global_addr | type=2
        v4u g0 = { 1u,
                   lbase + (unsigned)b * (unsigned)TDM_TILE_BYTES,
                   (unsigned)ga,
                   (unsigned)(ga >> 32) | 0x80000000u };
        // D# group1: data_size=2B, pad_enable, pad_interval=32dw(enc 4),
        // pad_amount=4dw(enc 3); tensor_dim0=192, tensor_dim1=32,
        // tile_dim0=192, tile_dim1=32, tensor_dim0_stride=192
        v8i g1 = { (int)0x07110000,
                   (int)(192u << 16),   // tensor_dim0[15:0] << 16
                   (int)(32u << 16),    // tensor_dim0[31:16]=0 | tensor_dim1[15:0]<<16
                   (int)(192u << 16),   // tensor_dim1[31:16]=0 | tile_dim0<<16
                   32,                  // tile_dim1 | tile_dim2=0
                   192,                 // tensor_dim0_stride[31:0]
                   0, 0 };
        v4i gz4 = { 0, 0, 0, 0 };
        v8i gz8 = { 0, 0, 0, 0, 0, 0, 0, 0 };
        __builtin_amdgcn_tensor_load_to_lds(g0, g1, gz4, gz4, gz8, 0);
    };

    float ssum[8] = {0.f, 0.f, 0.f, 0.f, 0.f, 0.f, 0.f, 0.f};

    if (wv == 0) {
        tdm_issue(t0, 0);
        __builtin_amdgcn_s_wait_tensorcnt(0);
    }
    __syncthreads();

    int buf = 0;
    for (int t = t0; t < t1; ++t) {
        const bool more = (t + 1 < t1);
        if (more && wv == 0) tdm_issue(t + 1, buf ^ 1);   // DMA overlaps compute

        v8f acc0 = {0.f, 0.f, 0.f, 0.f, 0.f, 0.f, 0.f, 0.f};
        v8f acc1 = {0.f, 0.f, 0.f, 0.f, 0.f, 0.f, 0.f, 0.f};
#pragma unroll
        for (int kk = 0; kk < 6; ++kk) {
            const int d   = 16 * kk + 8 * g;              // stored-dword offset in row
            const int off = 2 * d + 8 * (d >> 5);         // bf16 offset after TDM padding
            const __bf16* r0 = &Bsh[buf][ncol][off];
            const __bf16* r1 = &Bsh[buf][16 + ncol][off];
            V16 b0, b1;
            b0.h[0] = *(const v8bf*)r0;       b0.h[1] = *(const v8bf*)(r0 + 8);
            b1.h[0] = *(const v8bf*)r1;       b1.h[1] = *(const v8bf*)(r1 + 8);
            acc0 = __builtin_amdgcn_wmma_f32_16x16x32_bf16(
                false, afrag[kk].v, false, b0.v, (short)0, acc0, false, false);
            acc1 = __builtin_amdgcn_wmma_f32_16x16x32_bf16(
                false, afrag[kk].v, false, b1.v, (short)0, acc1, false, false);
        }
#pragma unroll
        for (int r = 0; r < 8; ++r)
            ssum[r] += __builtin_amdgcn_exp2f(acc0[r]) + __builtin_amdgcn_exp2f(acc1[r]);

        if (more) {
            if (wv == 0) __builtin_amdgcn_s_wait_tensorcnt(0);
            __syncthreads();
            buf ^= 1;
        }
    }

#pragma unroll
    for (int r = 0; r < 8; ++r) {
        float v = ssum[r];
        v += __shfl_xor(v, 1, 32);
        v += __shfl_xor(v, 2, 32);
        v += __shfl_xor(v, 4, 32);
        v += __shfl_xor(v, 8, 32);
        ssum[r] = v;
    }
    if (ncol == 0) {
        const int rowb = rowbase + 16 * wv + 8 * g;
#pragma unroll
        for (int r = 0; r < 8; ++r)
            partial[(size_t)bx * N_ROWS + rowb + r] = ssum[r];
    }
}

// ---------------------------------------------------------------------------
// Kernel 2b (fallback, small ws): stream f32 W, convert to bf16 in-kernel
// ---------------------------------------------------------------------------
__global__ __launch_bounds__(256) void gemm_lse_f32_kernel(
    const __bf16* __restrict__ A, const float* __restrict__ W,
    float* __restrict__ partial)
{
    __shared__ __bf16 Bsh[2][32][BSTRIDE];

    const int tid     = threadIdx.x;
    const int lane    = tid & 31;
    const int wv      = tid >> 5;
    const int g       = lane >> 4;
    const int ncol    = lane & 15;
    const int bx      = blockIdx.x;
    const int rowbase = blockIdx.y * M_TILE;

    V16 afrag[6];
    {
        const __bf16* Ar = A + (size_t)(rowbase + 16 * wv + ncol) * D_DIM;
#pragma unroll
        for (int kk = 0; kk < 6; ++kk) {
            afrag[kk].h[0] = *(const v8bf*)(Ar + kk * 32 + 16 * g);
            afrag[kk].h[1] = *(const v8bf*)(Ar + kk * 32 + 16 * g + 8);
        }
    }

    const float4* Wv4 = (const float4*)W;
    int sc[6], sq[6];
#pragma unroll
    for (int i = 0; i < 6; ++i) { int idx = i * 256 + tid; sc[i] = idx / 48; sq[i] = idx % 48; }

    float4 pf[6];
    auto prefetch = [&](int t) {
#pragma unroll
        for (int i = 0; i < 6; ++i)
            pf[i] = Wv4[(size_t)(t * 32 + sc[i]) * 48 + sq[i]];
    };
    auto stage = [&](int b) {
#pragma unroll
        for (int i = 0; i < 6; ++i) {
            v4bf o;
            o[0] = (__bf16)pf[i].x; o[1] = (__bf16)pf[i].y;
            o[2] = (__bf16)pf[i].z; o[3] = (__bf16)pf[i].w;
            *(v4bf*)&Bsh[b][sc[i]][4 * sq[i]] = o;
        }
    };

    const int t0 = (CT * bx) / NCHUNK;
    const int t1 = (CT * (bx + 1)) / NCHUNK;

    float ssum[8] = {0.f, 0.f, 0.f, 0.f, 0.f, 0.f, 0.f, 0.f};

    prefetch(t0);
    stage(0);
    __syncthreads();

    int buf = 0;
    for (int t = t0; t < t1; ++t) {
        const bool more = (t + 1 < t1);
        if (more) prefetch(t + 1);

        v8f acc0 = {0.f, 0.f, 0.f, 0.f, 0.f, 0.f, 0.f, 0.f};
        v8f acc1 = {0.f, 0.f, 0.f, 0.f, 0.f, 0.f, 0.f, 0.f};
#pragma unroll
        for (int kk = 0; kk < 6; ++kk) {
            V16 b0, b1;
            b0.h[0] = *(const v8bf*)&Bsh[buf][ncol][kk * 32 + 16 * g];
            b0.h[1] = *(const v8bf*)&Bsh[buf][ncol][kk * 32 + 16 * g + 8];
            b1.h[0] = *(const v8bf*)&Bsh[buf][16 + ncol][kk * 32 + 16 * g];
            b1.h[1] = *(const v8bf*)&Bsh[buf][16 + ncol][kk * 32 + 16 * g + 8];
            acc0 = __builtin_amdgcn_wmma_f32_16x16x32_bf16(
                false, afrag[kk].v, false, b0.v, (short)0, acc0, false, false);
            acc1 = __builtin_amdgcn_wmma_f32_16x16x32_bf16(
                false, afrag[kk].v, false, b1.v, (short)0, acc1, false, false);
        }
#pragma unroll
        for (int r = 0; r < 8; ++r)
            ssum[r] += __builtin_amdgcn_exp2f(acc0[r]) + __builtin_amdgcn_exp2f(acc1[r]);

        if (more) stage(buf ^ 1);
        __syncthreads();
        buf ^= 1;
    }

#pragma unroll
    for (int r = 0; r < 8; ++r) {
        float v = ssum[r];
        v += __shfl_xor(v, 1, 32);
        v += __shfl_xor(v, 2, 32);
        v += __shfl_xor(v, 4, 32);
        v += __shfl_xor(v, 8, 32);
        ssum[r] = v;
    }
    if (ncol == 0) {
        const int rowb = rowbase + 16 * wv + 8 * g;
#pragma unroll
        for (int r = 0; r < 8; ++r)
            partial[(size_t)bx * N_ROWS + rowb + r] = ssum[r];
    }
}

// ---------------------------------------------------------------------------
// Kernel 3: per-row margin swap + logsumexp -> per-row loss
// ---------------------------------------------------------------------------
__global__ __launch_bounds__(256) void finalize_kernel(
    const float* __restrict__ partial, const float* __restrict__ tgt,
    float* __restrict__ loss)
{
    const int n = blockIdx.x * 256 + threadIdx.x;
    float s = 0.f;
#pragma unroll 4
    for (int ch = 0; ch < NCHUNK; ++ch) s += partial[(size_t)ch * N_ROWS + n];

    const float t      = tgt[n];
    const float plain2 = K2F * t;
    const float numer2 = K2F * (t - 0.2f);
    s = s - __builtin_amdgcn_exp2f(plain2) + __builtin_amdgcn_exp2f(numer2);
    loss[n] = LN2F * (__builtin_amdgcn_logf(s) - numer2);   // v_log_f32 is log2
}

// ---------------------------------------------------------------------------
// Kernel 4: mean over 2048 rows -> scalar
// ---------------------------------------------------------------------------
__global__ __launch_bounds__(256) void reduce_kernel(
    const float* __restrict__ loss, float* __restrict__ out)
{
    __shared__ float sm[8];
    const int tid = threadIdx.x;
    float v = 0.f;
#pragma unroll
    for (int j = 0; j < 8; ++j) v += loss[tid + 256 * j];
#pragma unroll
    for (int m = 16; m >= 1; m >>= 1) v += __shfl_xor(v, m, 32);
    if ((tid & 31) == 0) sm[tid >> 5] = v;
    __syncthreads();
    if (tid == 0) {
        float tot = 0.f;
#pragma unroll
        for (int i = 0; i < 8; ++i) tot += sm[i];
        out[0] = tot * (1.0f / (float)N_ROWS);
    }
}

// ---------------------------------------------------------------------------
extern "C" void kernel_launch(void* const* d_in, const int* in_sizes, int n_in,
                              void* d_out, int out_size, void* d_ws, size_t ws_size,
                              hipStream_t stream)
{
    const float* x     = (const float*)d_in[0];
    const float* W     = (const float*)d_in[1];
    const int*   label = (const int*)d_in[2];
    float*       out   = (float*)d_out;

    char* ws = (char*)d_ws;
    __bf16* A       = (__bf16*)ws;                                  //   786432 B
    float*  tgt     = (float*)(ws + 786432);                        //     8192 B
    float*  partial = (float*)(ws + 786432 + 8192);                 //   262144 B
    float*  loss    = (float*)(ws + 786432 + 8192 + 262144);        //     8192 B
    __bf16* Wb      = (__bf16*)(ws + 1064960);                      // 38400000 B
    const size_t WS_BIG = 1064960u + (size_t)C_CLS * D_DIM * 2;

    normalize_kernel<<<N_ROWS / 8, 256, 0, stream>>>(x, W, label, A, tgt);

    if (ws_size >= WS_BIG) {
        // pre-convert W once, then TDM-fed GEMM (half the L2 streaming bytes)
        convert_w_kernel<<<(C_CLS * D_DIM / 4) / 256, 256, 0, stream>>>(W, Wb);
        gemm_lse_tdm_kernel<<<dim3(NCHUNK, N_ROWS / M_TILE), 256, 0, stream>>>(A, Wb, partial);
    } else {
        gemm_lse_f32_kernel<<<dim3(NCHUNK, N_ROWS / M_TILE), 256, 0, stream>>>(A, W, partial);
    }

    finalize_kernel<<<N_ROWS / 256, 256, 0, stream>>>(partial, tgt, loss);
    reduce_kernel<<<1, 256, 0, stream>>>(loss, out);
}
